// DeepSeekMoeWrapperBaseline_44418551775989
// MI455X (gfx1250) — matrix operand order repaired and checked by reading, not compiled, not run on hardware
//
#include <hip/hip_runtime.h>

// ---------------- problem constants ----------------
#define T_TOK 2048      // B*S tokens
#define DIM   2048      // HIDDEN
#define NEXP  8
#define FF    1408      // MOE_FF
#define SFF   2816      // SHARED_FF

// ---------------- GEMM tile config -----------------
#define BM 64
#define BN 128
#define BK 32
#define LDSB 40         // padded LDS row stride (bf16 elems): 80B -> conflict-free

typedef __attribute__((ext_vector_type(16))) __bf16 v16bf;
typedef __attribute__((ext_vector_type(8)))  float  v8f;

union FragBF { uint4 q[2]; v16bf v; };

__device__ __forceinline__ unsigned short f2bf(float f) {
  unsigned int u = __float_as_uint(f);
  u += 0x7FFFu + ((u >> 16) & 1u);        // round-to-nearest-even
  return (unsigned short)(u >> 16);
}
__device__ __forceinline__ unsigned int pack2(float a, float b) {
  return (unsigned int)f2bf(a) | ((unsigned int)f2bf(b) << 16);
}

// CDNA5 async copy: 16B global -> LDS per lane, tracked by ASYNCcnt
__device__ __forceinline__ void async_copy16(unsigned lds_byte_addr, const void* gaddr) {
  asm volatile("global_load_async_to_lds_b128 %0, %1, off"
               :: "v"(lds_byte_addr), "v"((unsigned long long)gaddr)
               : "memory");
}
__device__ __forceinline__ void wait_async0() {
  asm volatile("s_wait_asynccnt 0" ::: "memory");
}

// ---------------------------------------------------------------------------
// Kernel 0: zero per-expert counters
// ---------------------------------------------------------------------------
__global__ void init_counts(int* counts) {
  if (threadIdx.x < NEXP) counts[threadIdx.x] = 0;
}

// ---------------------------------------------------------------------------
// Kernel: fp32 -> bf16 bulk convert (8 elems / thread, exact grid)
// ---------------------------------------------------------------------------
__global__ __launch_bounds__(256)
void convert_bf16_kernel(const float* __restrict__ src, unsigned short* __restrict__ dst) {
  size_t i = (size_t)blockIdx.x * 256 + threadIdx.x;
  const float4* s = (const float4*)src;
  float4 a = s[2 * i], b = s[2 * i + 1];
  uint4 o;
  o.x = pack2(a.x, a.y); o.y = pack2(a.z, a.w);
  o.z = pack2(b.x, b.y); o.w = pack2(b.z, b.w);
  ((uint4*)dst)[i] = o;
}

// ---------------------------------------------------------------------------
// Kernel 1: gating (sigmoid + top-2) + x -> bf16 conversion + expert lists
// one block (256 thr) per token
// ---------------------------------------------------------------------------
__global__ __launch_bounds__(256)
void gate_kernel(const float* __restrict__ x, const float* __restrict__ gw,
                 unsigned short* __restrict__ xb, float* __restrict__ tkw,
                 int* __restrict__ lists, int* __restrict__ counts)
{
  __shared__ float red[NEXP * 256];
  const int t = blockIdx.x, tid = threadIdx.x;

  float acc[NEXP];
#pragma unroll
  for (int e = 0; e < NEXP; e++) acc[e] = 0.f;

  for (int i = tid; i < DIM; i += 256) {
    float v = x[(size_t)t * DIM + i];
    xb[(size_t)t * DIM + i] = f2bf(v);
#pragma unroll
    for (int e = 0; e < NEXP; e++) acc[e] += v * gw[e * DIM + i];
  }
#pragma unroll
  for (int e = 0; e < NEXP; e++) red[e * 256 + tid] = acc[e];
  __syncthreads();
  for (int s = 128; s > 0; s >>= 1) {
    if (tid < s) {
#pragma unroll
      for (int e = 0; e < NEXP; e++) red[e * 256 + tid] += red[e * 256 + tid + s];
    }
    __syncthreads();
  }
  if (tid == 0) {
    float sc[NEXP];
#pragma unroll
    for (int e = 0; e < NEXP; e++) sc[e] = 1.f / (1.f + expf(-red[e * 256]));
    int i0 = 0;
    for (int e = 1; e < NEXP; e++) if (sc[e] > sc[i0]) i0 = e;
    int i1 = (i0 == 0) ? 1 : 0;
    for (int e = 0; e < NEXP; e++) if (e != i0 && sc[e] > sc[i1]) i1 = e;
    float w0 = sc[i0], w1 = sc[i1];
    float s = 2.5f / (w0 + w1 + 1e-20f);
    tkw[t * 2 + 0] = w0 * s;
    tkw[t * 2 + 1] = w1 * s;
    int s0 = atomicAdd(&counts[i0], 1);
    lists[i0 * T_TOK + s0] = t;                 // k = 0
    int s1 = atomicAdd(&counts[i1], 1);
    lists[i1 * T_TOK + s1] = t | (1 << 16);     // k = 1
  }
}

// ---------------------------------------------------------------------------
// WMMA GEMM core:  C(MxN) = A(bf16, lda) @ B(bf16, F x K)^T
//   GATED  : dual-B (gate/up), epilogue silu(g)*u -> bf16 out
//   GATHER : A rows gathered through token list (routed stage 1)
//   SCATTER: fp32 out scattered to per-(token,k) slots (routed stage 2)
// 256 threads = 8 waves; each wave owns a 32x32 patch of the 64x128 tile.
// Double-buffered: ALL tiles staged via global_load_async_to_lds_b128; stage
// s+1's async DMA overlaps stage s's ds_load + v_wmma compute.
// ---------------------------------------------------------------------------
template<int GATED, int GATHER, int SCATTER>
__global__ __launch_bounds__(256)
void moe_gemm(const unsigned short* __restrict__ A, int lda,
              const unsigned short* __restrict__ B0w,
              const unsigned short* __restrict__ B1w, int K,
              void* __restrict__ OutP, int ldo,
              const int* __restrict__ list, const int* __restrict__ cntPtr)
{
  __shared__ unsigned short sA [2 * BM * LDSB];
  __shared__ unsigned short sB0[2 * BN * LDSB];
  __shared__ unsigned short sB1[GATED ? (2 * BN * LDSB) : 1];
  __shared__ int sTok[BM];

  const int tid = threadIdx.x;
  const int m0  = blockIdx.y * BM;
  const int n0  = blockIdx.x * BN;

  int cnt = 0x7fffffff;
  if (GATHER || SCATTER) {
    cnt = *cntPtr;
    if (m0 >= cnt) return;                       // uniform early-exit
    if (tid < BM) {
      int slot = m0 + tid;
      sTok[tid] = (slot < cnt) ? list[slot] : -1;
    }
  }

  v8f c0[2][2], c1[2][2];
#pragma unroll
  for (int a = 0; a < 2; a++)
#pragma unroll
    for (int b = 0; b < 2; b++)
#pragma unroll
      for (int i = 0; i < 8; i++) { c0[a][b][i] = 0.f; if (GATED) c1[a][b][i] = 0.f; }

  const int arow = tid >> 2, achk = tid & 3;     // A staging: 64 rows x 4 chunks(16B)
  const int brow = tid >> 1, bhlf = tid & 1;     // B staging: 128 rows x 2 chunks(32B)

  const int lane = tid & 31;
  const int wv   = tid >> 5;
  const int wm   = (wv >> 2) * 32;               // wave row base (0/32)
  const int wn   = (wv & 3) * 32;                // wave col base (0..96)
  const int lr   = lane & 15;
  const int lh   = lane >> 4;

  // per-thread invariant LDS byte addresses (buffer 0)
  const unsigned aDst = (unsigned)(uintptr_t)(sA  + arow * LDSB + achk * 8);
  const unsigned bDst = (unsigned)(uintptr_t)(sB0 + brow * LDSB + bhlf * 16);
  const unsigned bDst1 = GATED ? (unsigned)(uintptr_t)(sB1 + brow * LDSB + bhlf * 16) : 0u;

  int srowA = m0 + arow;                          // A source row (gather-resolved below)

  const int nsteps = K / BK;

  // issue stage s's async copies into buffer (s & 1)
  auto stage = [&](int s) {
    const int kb  = s * BK;
    const unsigned abuf = (unsigned)(s & 1) * (BM * LDSB * 2);
    const unsigned bbuf = (unsigned)(s & 1) * (BN * LDSB * 2);
    async_copy16(aDst + abuf, A + (size_t)srowA * lda + kb + achk * 8);
    size_t base = (size_t)(n0 + brow) * (size_t)K + kb + bhlf * 16;
    async_copy16(bDst + bbuf,      B0w + base);
    async_copy16(bDst + bbuf + 16, B0w + base + 8);
    if (GATED) {
      async_copy16(bDst1 + bbuf,      B1w + base);
      async_copy16(bDst1 + bbuf + 16, B1w + base + 8);
    }
    if (kb + BK < K) {                            // prefetch one stage further out
      __builtin_prefetch(B0w + base + BK, 0, 0);
      if (GATED) __builtin_prefetch(B1w + base + BK, 0, 0);
    }
  };

  __syncthreads();                                // sTok visible
  if (GATHER) { int pk = sTok[arow]; srowA = (pk >= 0) ? (pk & 0xFFFF) : 0; }
  stage(0);

  for (int s = 0; s < nsteps; s++) {
    wait_async0();            // my stage-s copies done (stage s+1 not yet issued)
    __syncthreads();          // everyone's stage-s data in LDS; prior reads of
                              // buffer (s+1)&1 finished -> safe to overwrite
    if (s + 1 < nsteps) stage(s + 1);   // DMA overlaps compute below

    const int buf = s & 1;
    const unsigned short* sAb  = sA  + buf * (BM * LDSB);
    const unsigned short* sB0b = sB0 + buf * (BN * LDSB);

    // ---- load fragments, contiguous k order (k = lh*16 + s) for A and B ----
    FragBF afr[2];
#pragma unroll
    for (int im = 0; im < 2; im++) {
      const unsigned short* pa = sAb + (wm + im * 16 + lr) * LDSB;
      afr[im].q[0] = *(const uint4*)(pa + lh * 16);       // K = lh*16 + 0..7
      afr[im].q[1] = *(const uint4*)(pa + lh * 16 + 8);   // K = lh*16 + 8..15
    }
    FragBF b0f[2], b1f[2];
#pragma unroll
    for (int in = 0; in < 2; in++) {
      const unsigned short* pb = sB0b + (wn + in * 16 + lr) * LDSB + lh * 16;
      b0f[in].q[0] = *(const uint4*)(pb);                 // K = lh*16 + 0..7
      b0f[in].q[1] = *(const uint4*)(pb + 8);             // K = lh*16 + 8..15
      if (GATED) {
        const unsigned short* pc =
            sB1 + buf * (BN * LDSB) + (wn + in * 16 + lr) * LDSB + lh * 16;
        b1f[in].q[0] = *(const uint4*)(pc);
        b1f[in].q[1] = *(const uint4*)(pc + 8);
      }
    }
#pragma unroll
    for (int im = 0; im < 2; im++)
#pragma unroll
      for (int in = 0; in < 2; in++) {
        c0[im][in] = __builtin_amdgcn_wmma_f32_16x16x32_bf16(
            false, afr[im].v, false, b0f[in].v, (short)0, c0[im][in], false, false);
        if (GATED)
          c1[im][in] = __builtin_amdgcn_wmma_f32_16x16x32_bf16(
              false, afr[im].v, false, b1f[in].v, (short)0, c1[im][in], false, false);
      }
  }

  // ---- epilogue ----
#pragma unroll
  for (int im = 0; im < 2; im++)
#pragma unroll
    for (int in = 0; in < 2; in++)
#pragma unroll
      for (int r = 0; r < 8; r++) {
        int lm = wm + im * 16 + lh * 8 + r;      // local row 0..63
        int gn = n0 + wn + in * 16 + lr;         // global col
        if (GATED) {
          float g = c0[im][in][r];
          float u = c1[im][in][r];
          float h = g / (1.f + expf(-g)) * u;    // silu(g) * u
          if (!GATHER || (m0 + lm) < cnt)
            ((unsigned short*)OutP)[(size_t)(m0 + lm) * ldo + gn] = f2bf(h);
        } else if (SCATTER) {
          int pk = sTok[lm];
          if (pk >= 0)
            ((float*)OutP)[(size_t)((pk & 0xFFFF) * 2 + (pk >> 16)) * ldo + gn] =
                c0[im][in][r];
        } else {
          ((float*)OutP)[(size_t)(m0 + lm) * ldo + gn] = c0[im][in][r];
        }
      }
}

// ---------------------------------------------------------------------------
// Final: out += w0*pick[t][0] + w1*pick[t][1]   (out already = shared_out)
// ---------------------------------------------------------------------------
__global__ __launch_bounds__(256)
void combine_kernel(const float* __restrict__ pick, const float* __restrict__ tkw,
                    float* __restrict__ out)
{
  size_t gid = (size_t)blockIdx.x * 256 + threadIdx.x;   // float4 index
  int t  = (int)(gid / (DIM / 4));
  int c4 = (int)(gid % (DIM / 4));
  const float4 p0 = ((const float4*)(pick + (size_t)(t * 2)     * DIM))[c4];
  const float4 p1 = ((const float4*)(pick + (size_t)(t * 2 + 1) * DIM))[c4];
  float w0 = tkw[t * 2], w1 = tkw[t * 2 + 1];
  float4* po = ((float4*)(out + (size_t)t * DIM)) + c4;
  float4 v = *po;
  v.x += w0 * p0.x + w1 * p1.x;
  v.y += w0 * p0.y + w1 * p1.y;
  v.z += w0 * p0.z + w1 * p1.z;
  v.w += w0 * p0.w + w1 * p1.w;
  *po = v;
}

// ---------------------------------------------------------------------------
extern "C" void kernel_launch(void* const* d_in, const int* in_sizes, int n_in,
                              void* d_out, int out_size, void* d_ws, size_t ws_size,
                              hipStream_t stream) {
  (void)in_sizes; (void)n_in; (void)out_size; (void)ws_size;
  const float* x         = (const float*)d_in[0];
  const float* gate_w    = (const float*)d_in[1];
  const float* shared_wg = (const float*)d_in[2];
  const float* shared_wu = (const float*)d_in[3];
  const float* shared_wd = (const float*)d_in[4];
  const float* exp_wg    = (const float*)d_in[5];
  const float* exp_wu    = (const float*)d_in[6];
  const float* exp_wd    = (const float*)d_in[7];
  float* out = (float*)d_out;

  char* ws = (char*)d_ws;
  size_t o = 0;
  unsigned short* xb  = (unsigned short*)(ws + o); o += (size_t)T_TOK * DIM * 2;   // 8 MB
  unsigned short* hs  = (unsigned short*)(ws + o); o += (size_t)T_TOK * SFF * 2;   // 11.5 MB
  unsigned short* hrt = (unsigned short*)(ws + o); o += (size_t)T_TOK * FF  * 2;   // 5.8 MB (reused per expert)
  float* pick   = (float*)(ws + o); o += (size_t)T_TOK * 2 * DIM * 4;              // 33.6 MB
  float* tkw    = (float*)(ws + o); o += (size_t)T_TOK * 2 * 4;
  int*   lists  = (int*)(ws + o);   o += (size_t)NEXP * T_TOK * 4;
  int*   counts = (int*)(ws + o);   o += 256;
  // bf16 weight copies: shared (persistent this call) + per-expert reuse buffer
  unsigned short* wgS = (unsigned short*)(ws + o); o += (size_t)SFF * DIM * 2;     // 11.5 MB
  unsigned short* wuS = (unsigned short*)(ws + o); o += (size_t)SFF * DIM * 2;     // 11.5 MB
  unsigned short* wdS = (unsigned short*)(ws + o); o += (size_t)DIM * SFF * 2;     // 11.5 MB
  unsigned short* wgE = (unsigned short*)(ws + o); o += (size_t)FF * DIM * 2;      // 5.8 MB
  unsigned short* wuE = (unsigned short*)(ws + o); o += (size_t)FF * DIM * 2;      // 5.8 MB
  unsigned short* wdE = (unsigned short*)(ws + o); o += (size_t)DIM * FF * 2;      // 5.8 MB

  const int gS = (int)(((size_t)SFF * DIM / 8) / 256);   // 2816 blocks
  const int gE = (int)(((size_t)FF  * DIM / 8) / 256);   // 1408 blocks

  init_counts<<<1, 32, 0, stream>>>(counts);
  gate_kernel<<<T_TOK, 256, 0, stream>>>(x, gate_w, xb, tkw, lists, counts);

  // shared weights -> bf16 (once per call)
  convert_bf16_kernel<<<gS, 256, 0, stream>>>(shared_wg, wgS);
  convert_bf16_kernel<<<gS, 256, 0, stream>>>(shared_wu, wuS);
  convert_bf16_kernel<<<gS, 256, 0, stream>>>(shared_wd, wdS);

  // shared expert: stage 1 (gated) -> hs(bf16), stage 2 -> d_out (fp32)
  moe_gemm<1, 0, 0><<<dim3(SFF / BN, T_TOK / BM), 256, 0, stream>>>(
      xb, DIM, wgS, wuS, DIM, hs, SFF, nullptr, nullptr);
  moe_gemm<0, 0, 0><<<dim3(DIM / BN, T_TOK / BM), 256, 0, stream>>>(
      hs, SFF, wdS, nullptr, SFF, out, DIM, nullptr, nullptr);

  // routed experts: convert -> gather stage-1 -> hrt, stage-2 scatter -> pick
  for (int e = 0; e < NEXP; e++) {
    convert_bf16_kernel<<<gE, 256, 0, stream>>>(exp_wg + (size_t)e * FF * DIM, wgE);
    convert_bf16_kernel<<<gE, 256, 0, stream>>>(exp_wu + (size_t)e * FF * DIM, wuE);
    convert_bf16_kernel<<<gE, 256, 0, stream>>>(exp_wd + (size_t)e * DIM * FF, wdE);
    moe_gemm<1, 1, 0><<<dim3(FF / BN, T_TOK / BM), 256, 0, stream>>>(
        xb, DIM, wgE, wuE, DIM, hrt, FF, lists + e * T_TOK, counts + e);
    moe_gemm<0, 0, 1><<<dim3(DIM / BN, T_TOK / BM), 256, 0, stream>>>(
        hrt, FF, wdE, nullptr, FF, pick, DIM, lists + e * T_TOK, counts + e);
  }

  combine_kernel<<<(T_TOK * (DIM / 4)) / 256, 256, 0, stream>>>(pick, tkw, out);
}
